// BasicBlock_quant_11751030522176
// MI455X (gfx1250) — 8-bit matrix operand order repaired and checked by reading, compiled, not run on hardware
//
#include <hip/hip_runtime.h>
#include <hip/hip_bf16.h>
#include <stdint.h>


typedef int   v16i __attribute__((ext_vector_type(16)));
typedef float v8f  __attribute__((ext_vector_type(8)));

#define EPS 1e-5f

#define NB      64
#define CIN     128
#define HIN     56
#define WIN     56
#define CPL     256
#define HOUT    28
#define WOUT    28
#define HWIN    (HIN*WIN)
#define HWOUT   (HOUT*WOUT)
#define NPIX    (NB*HWOUT)


__device__ __forceinline__ uint8_t sgnfp8(float v) { return v >= 0.f ? (uint8_t)0x38 : (uint8_t)0xB8; }

__global__ __launch_bounds__(256) void binarize_x_kernel(const float* __restrict__ x,
                                                         uint8_t* __restrict__ xq) {
    int idx = blockIdx.x * 256 + threadIdx.x;
    int total = NB * HWIN * CIN;
    if (idx >= total) return;
    int c   = idx & (CIN - 1);
    int rst = idx >> 7;
    int pix = rst % HWIN;
    int b   = rst / HWIN;
    xq[idx] = sgnfp8(x[(b * CIN + c) * HWIN + pix]);
}

__global__ __launch_bounds__(256) void binarize_w_kernel(const float* __restrict__ w,
                                                         uint8_t* __restrict__ wq,
                                                         int O, int I, int KS) {
    int K = KS * KS * I;
    int total = O * K;
    int idx = blockIdx.x * 256 + threadIdx.x;
    if (idx >= total) return;
    int o = idx / K;
    int k = idx % K;
    int khkw = k / I;
    int c    = k % I;
    int kh = khkw / KS, kw = khkw % KS;
    wq[idx] = sgnfp8(w[((o * I + c) * KS + kh) * KS + kw]);
}

__global__ __launch_bounds__(256) void absmean_kernel(const float* __restrict__ w, int n,
                                                      float* __restrict__ alpha) {
    __shared__ float red[256];
    float s = 0.f;
    for (int i = threadIdx.x; i < n; i += 256) s += fabsf(w[i]);
    red[threadIdx.x] = s;
    __syncthreads();
    for (int off = 128; off > 0; off >>= 1) {
        if (threadIdx.x < off) red[threadIdx.x] += red[threadIdx.x + off];
        __syncthreads();
    }
    if (threadIdx.x == 0) *alpha = red[0] / (float)n;
}


template <int KS, int STRIDE, int PAD, int CIN_T, int HIN_T, int WIN_T, bool ACC>
__global__ __launch_bounds__(256, 1) void bgemm_conv_kernel(const uint8_t* __restrict__ xq,
                                                            const uint8_t* __restrict__ wq,
                                                            const float* __restrict__ alpha_p,
                                                            float* __restrict__ out) {
    constexpr int KTOT = KS * KS * CIN_T;
    constexpr int S    = KTOT / 128;

    __shared__ uint8_t ldsA[3][128 * 128];
    __shared__ uint8_t ldsB[2][64 * 128];

    const int t    = threadIdx.x;
    const int wave = t >> 5;
    const int lane = t & 31;
    const int half = lane >> 4;
    const int lm   = lane & 15;

    const int mbase = blockIdx.y * 128;
    const int pbase = blockIdx.x * 64;

    const int jB    = t >> 2;
    const int boffB = (t & 3) * 32;
    const int pj  = pbase + jB;
    const int bB  = pj / HWOUT;
    const int prm = pj % HWOUT;
    const int oh  = prm / WOUT;
    const int ow  = prm % WOUT;

    const int mArow = t >> 1;
    const int aoffA = (t & 1) * 64;
    const uint8_t* arow = wq + (size_t)(mbase + mArow) * KTOT + aoffA;

    auto load_A = [&](int s, int4 va[4]) {
        const uint8_t* asrc = arow + s * 128;
#pragma unroll
        for (int i = 0; i < 4; ++i) va[i] = *(const int4*)(asrc + 16 * i);
        if (s * 128 + 128 < KTOT) __builtin_prefetch(asrc + 128, 0, 1);
    };
    auto store_A = [&](int buf, const int4 va[4]) {
#pragma unroll
        for (int i = 0; i < 4; ++i)
            *(int4*)&ldsA[buf][mArow * 128 + aoffA + 16 * i] = va[i];
    };
    auto load_B = [&](int s, int4 vb[2]) {
        const int k0   = s * 128;
        const int khkw = k0 / CIN_T;
        const int c0   = k0 % CIN_T;
        const int kh   = khkw / KS;
        const int kw   = khkw % KS;
        const int ih   = oh * STRIDE - PAD + kh;
        const int iw   = ow * STRIDE - PAD + kw;
        vb[0] = make_int4(0, 0, 0, 0);
        vb[1] = make_int4(0, 0, 0, 0);
        if (ih >= 0 && ih < HIN_T && iw >= 0 && iw < WIN_T) {
            const uint8_t* src = xq +
                ((size_t)((bB * HIN_T + ih) * WIN_T + iw) * CIN_T + c0 + boffB);
            vb[0] = *(const int4*)src;
            vb[1] = *(const int4*)(src + 16);
        }
    };
    auto store_B = [&](int buf, const int4 vb[2]) {
        *(int4*)&ldsB[buf][jB * 128 + boffB]      = vb[0];
        *(int4*)&ldsB[buf][jB * 128 + boffB + 16] = vb[1];
    };

    v8f acc[4] = {};

    {
        int4 va[4], vb[2];
        load_A(0, va);
        load_B(0, vb);
        store_A(0, va);
        store_B(0, vb);
    }
    __syncthreads();

    int ar = 0, aw = 1, br = 0;
    for (int s = 0; s < S; ++s) {
        const bool have_next = (s + 1 < S);
        int4 nb[2];
        if (have_next) {
            int4 va[4];
            load_A(s + 1, va);
            store_A(aw, va);
            load_B(s + 1, nb);
        }

        const int* lA = (const int*)ldsA[ar];
        const int  mrow = wave * 16 + lm;
        v16i afrag;
#pragma unroll
        for (int v = 0; v < 16; ++v) {
            const int v7 = v & 7;
            const int kb = ((v >> 2) << 5) + (half << 4) + ((v & 3) << 2);
            afrag[v] = lA[mrow * 32 + (kb >> 2)];
        }

        const int* lB = (const int*)ldsB[br];
#pragma unroll
        for (int nn = 0; nn < 4; ++nn) {
            const int j = nn * 16 + lm;
            v16i bfrag;
#pragma unroll
            for (int v = 0; v < 16; ++v) {
                const int kb = ((v >> 2) << 5) + (half << 4) + ((v & 3) << 2);
                bfrag[v] = lB[j * 32 + (kb >> 2)];
            }
            acc[nn] = __builtin_amdgcn_wmma_f32_16x16x128_fp8_fp8(
                afrag, bfrag, (short)0, acc[nn], false, false);
        }

        if (have_next) {
            __syncthreads();
            store_B(br ^ 1, nb);
            __syncthreads();
        }
        ar = (ar == 2) ? 0 : ar + 1;
        aw = (aw == 2) ? 0 : aw + 1;
        br ^= 1;
    }

    const float alpha = *alpha_p;
#pragma unroll
    for (int nn = 0; nn < 4; ++nn) {
        const int pcol = pbase + nn * 16 + lm;
        const int b    = pcol / HWOUT;
        const int pix  = pcol % HWOUT;
#pragma unroll
        for (int r = 0; r < 8; ++r) {
            const int m = mbase + wave * 16 + half * 8 + r;
            const size_t idx = (size_t)(b * CPL + m) * HWOUT + pix;
            const float val = acc[nn][r] * alpha;
            if (ACC) out[idx] += val;
            else     out[idx]  = val;
        }
    }
}


__global__ __launch_bounds__(256) void bn_stats_kernel(const float* __restrict__ buf,
                                                       float* __restrict__ mean,
                                                       float* __restrict__ invstd) {
    const int c = blockIdx.x;
    __shared__ float rs[256], rq[256];
    float s = 0.f, q = 0.f;
    for (int b = 0; b < NB; ++b) {
        const float* base = buf + (size_t)(b * CPL + c) * HWOUT;
        for (int i = threadIdx.x; i < HWOUT; i += 256) {
            float v = base[i];
            s += v; q += v * v;
        }
    }
    rs[threadIdx.x] = s; rq[threadIdx.x] = q;
    __syncthreads();
    for (int off = 128; off > 0; off >>= 1) {
        if (threadIdx.x < off) {
            rs[threadIdx.x] += rs[threadIdx.x + off];
            rq[threadIdx.x] += rq[threadIdx.x + off];
        }
        __syncthreads();
    }
    if (threadIdx.x == 0) {
        const float cnt = (float)(NB * HWOUT);
        float m = rs[0] / cnt;
        float var = rq[0] / cnt - m * m;
        mean[c] = m;
        invstd[c] = rsqrtf(var + EPS);
    }
}

__global__ __launch_bounds__(256) void bn1_sign_kernel(const float* __restrict__ buf,
                                                       const float* __restrict__ mean,
                                                       const float* __restrict__ invstd,
                                                       const float* __restrict__ g,
                                                       const float* __restrict__ bta,
                                                       uint8_t* __restrict__ xq2) {
    int idx = blockIdx.x * 256 + threadIdx.x;
    int total = NB * HWOUT * CPL;
    if (idx >= total) return;
    int c   = idx & (CPL - 1);
    int rst = idx >> 8;
    int pix = rst % HWOUT;
    int b   = rst / HWOUT;
    float v = buf[(size_t)(b * CPL + c) * HWOUT + pix];
    v = (v - mean[c]) * invstd[c] * g[c] + bta[c];
    xq2[idx] = sgnfp8(v);
}

__global__ __launch_bounds__(256) void bn2_out_kernel(const float* __restrict__ buf,
                                                      const float* __restrict__ mean,
                                                      const float* __restrict__ invstd,
                                                      const float* __restrict__ g,
                                                      const float* __restrict__ bta,
                                                      float* __restrict__ out) {
    int idx = blockIdx.x * 256 + threadIdx.x;
    int total = NB * CPL * HWOUT;
    if (idx >= total) return;
    int c = (idx / HWOUT) & (CPL - 1);
    out[idx] = (buf[idx] - mean[c]) * invstd[c] * g[c] + bta[c];
}


extern "C" void kernel_launch(void* const* d_in, const int* in_sizes, int n_in,
                              void* d_out, int out_size, void* d_ws, size_t ws_size,
                              hipStream_t stream) {
    (void)in_sizes; (void)n_in; (void)out_size; (void)ws_size;

    const float* x  = (const float*)d_in[0];
    const float* w1 = (const float*)d_in[1];
    const float* g1 = (const float*)d_in[2];
    const float* b1 = (const float*)d_in[3];
    const float* w2 = (const float*)d_in[4];
    const float* g2 = (const float*)d_in[5];
    const float* b2 = (const float*)d_in[6];
    const float* ws = (const float*)d_in[7];
    float* out = (float*)d_out;

    char* base = (char*)d_ws;
    size_t off = 0;
    auto carve = [&](size_t bytes) -> char* {
        char* p = base + off;
        off = (off + bytes + 255) & ~(size_t)255;
        return p;
    };
    uint8_t* xq    = (uint8_t*)carve((size_t)NB * HWIN * CIN);
    uint8_t* xq2   = (uint8_t*)carve((size_t)NB * HWOUT * CPL);
    uint8_t* wq1   = (uint8_t*)carve((size_t)CPL * 9 * CIN);
    uint8_t* wq2   = (uint8_t*)carve((size_t)CPL * 9 * CPL);
    uint8_t* wqs   = (uint8_t*)carve((size_t)CPL * CIN);
    float*   c1buf = (float*)carve((size_t)NB * CPL * HWOUT * 4);
    float*   ybuf  = (float*)carve((size_t)NB * CPL * HWOUT * 4);
    float*   a1    = (float*)carve(256);
    float*   a2    = (float*)carve(256);
    float*   as    = (float*)carve(256);
    float*   mean1 = (float*)carve(CPL * 4);
    float*   istd1 = (float*)carve(CPL * 4);
    float*   mean2 = (float*)carve(CPL * 4);
    float*   istd2 = (float*)carve(CPL * 4);

    {
        int total = NB * HWIN * CIN;
        binarize_x_kernel<<<(total + 255) / 256, 256, 0, stream>>>(x, xq);
    }
    {
        int n1 = CPL * CIN * 9, n2 = CPL * CPL * 9, nsw = CPL * CIN;
        binarize_w_kernel<<<(n1 + 255) / 256, 256, 0, stream>>>(w1, wq1, CPL, CIN, 3);
        binarize_w_kernel<<<(n2 + 255) / 256, 256, 0, stream>>>(w2, wq2, CPL, CPL, 3);
        binarize_w_kernel<<<(nsw + 255) / 256, 256, 0, stream>>>(ws, wqs, CPL, CIN, 1);
        absmean_kernel<<<1, 256, 0, stream>>>(w1, n1, a1);
        absmean_kernel<<<1, 256, 0, stream>>>(w2, n2, a2);
        absmean_kernel<<<1, 256, 0, stream>>>(ws, nsw, as);
    }

    const dim3 gGemm(NPIX / 64, CPL / 128);

    bgemm_conv_kernel<3, 2, 1, CIN, HIN, WIN, false>
        <<<gGemm, 256, 0, stream>>>(xq, wq1, a1, c1buf);
    bn_stats_kernel<<<CPL, 256, 0, stream>>>(c1buf, mean1, istd1);
    {
        int total = NB * HWOUT * CPL;
        bn1_sign_kernel<<<(total + 255) / 256, 256, 0, stream>>>(c1buf, mean1, istd1, g1, b1, xq2);
    }
    bgemm_conv_kernel<3, 1, 1, CPL, HOUT, WOUT, false>
        <<<gGemm, 256, 0, stream>>>(xq2, wq2, a2, ybuf);
    bgemm_conv_kernel<1, 2, 0, CIN, HIN, WIN, true>
        <<<gGemm, 256, 0, stream>>>(xq, wqs, as, ybuf);
    bn_stats_kernel<<<CPL, 256, 0, stream>>>(ybuf, mean2, istd2);
    {
        int total = NB * CPL * HWOUT;
        bn2_out_kernel<<<(total + 255) / 256, 256, 0, stream>>>(ybuf, mean2, istd2, g2, b2, out);
    }
}
